// LittleBitITQSpecLinear_8615704396412
// MI455X (gfx1250) — matrix operand order repaired and checked by reading, not compiled, not run on hardware
//
#include <hip/hip_runtime.h>
#include <hip/hip_bf16.h>

typedef _Float16 v16h __attribute__((ext_vector_type(16)));
typedef _Float16 v8h  __attribute__((ext_vector_type(8)));
typedef _Float16 v4h  __attribute__((ext_vector_type(4)));
typedef float    v8f  __attribute__((ext_vector_type(8)));
typedef int      v4i  __attribute__((ext_vector_type(4)));

#define IN_F  4096
#define OUT_F 4096
#define SPLIT 1024
#define NTOT  8192

#define BM 128
#define BN 128
#define BK 32
#define BKP 40   // padded LDS row stride in f16 elems (80B -> conflict-free b128)

// ---------------------------------------------------------------------------
// gfx1250 async global->LDS copy (ASYNCcnt) with sync fallback
// ---------------------------------------------------------------------------
#if __has_builtin(__builtin_amdgcn_global_load_async_to_lds_b128)
#define ASYNC_LDS 1
#else
#define ASYNC_LDS 0
#endif

typedef __attribute__((address_space(1))) v4i as1_v4i;   // global
typedef __attribute__((address_space(3))) v4i as3_v4i;   // LDS

__device__ __forceinline__ void cp_b128(const _Float16* g, _Float16* l) {
#if ASYNC_LDS
  __builtin_amdgcn_global_load_async_to_lds_b128((as1_v4i*)g, (as3_v4i*)l, 0, 0);
#else
  *reinterpret_cast<v8h*>(l) = *reinterpret_cast<const v8h*>(g);
#endif
}

__device__ __forceinline__ void wait_async0() {
#if ASYNC_LDS
#if __has_builtin(__builtin_amdgcn_s_wait_asynccnt)
  __builtin_amdgcn_s_wait_asynccnt(0);
#else
  asm volatile("s_wait_asynccnt 0x0" ::: "memory");
#endif
#endif
}

__device__ __forceinline__ void wait_ds0() {
  asm volatile("s_wait_dscnt 0x0" ::: "memory");
}

// ---------------------------------------------------------------------------
// sign(W) -> f16 (+1 / -1 / 0), natural layout (which is exactly B^T for WMMA)
// ---------------------------------------------------------------------------
__global__ __launch_bounds__(256) void pack_sign_f16(const float* __restrict__ src,
                                                     _Float16* __restrict__ dst, int n4) {
  int i = blockIdx.x * blockDim.x + threadIdx.x;
  if (i >= n4) return;
  float4 w = reinterpret_cast<const float4*>(src)[i];
  v4h o;
  o.x = (_Float16)((w.x > 0.f) ? 1.f : (w.x < 0.f ? -1.f : 0.f));
  o.y = (_Float16)((w.y > 0.f) ? 1.f : (w.y < 0.f ? -1.f : 0.f));
  o.z = (_Float16)((w.z > 0.f) ? 1.f : (w.z < 0.f ? -1.f : 0.f));
  o.w = (_Float16)((w.w > 0.f) ? 1.f : (w.w < 0.f ? -1.f : 0.f));
  reinterpret_cast<v4h*>(dst)[i] = o;
}

// ---------------------------------------------------------------------------
// GEMM1: H[n,s] = (sum_k f16(x[n,k]*v2[k]) * sign(V[s,k])) * v1[s]*u2[s]
// blockIdx.z selects primary / residual path. Double-buffered LDS, 1 barrier/iter.
// ---------------------------------------------------------------------------
__global__ __launch_bounds__(256) void gemm1_kernel(
    const float* __restrict__ x,
    const _Float16* __restrict__ Vp, const _Float16* __restrict__ Vr,
    const float* __restrict__ v2p, const float* __restrict__ v2r,
    const float* __restrict__ v1p, const float* __restrict__ v1r,
    const float* __restrict__ u2p, const float* __restrict__ u2r,
    _Float16* __restrict__ Hp, _Float16* __restrict__ Hr)
{
  __shared__ _Float16 As[2][BM][BKP];
  __shared__ _Float16 Bs[2][BN][BKP];

  const int t    = threadIdx.x;
  const int lane = t & 31;
  const int wave = t >> 5;
  const int hlf  = lane >> 4;     // half-wave: selects K-subblock per ISA layout
  const int lr   = lane & 15;
  const int waveM = wave & 3;     // 4 waves along M (32 rows each)
  const int waveN = wave >> 2;    // 2 waves along N (64 cols each)
  const int m0 = blockIdx.y * BM;
  const int s0 = blockIdx.x * BN;
  const int path = blockIdx.z;

  const _Float16* W  = path ? Vr  : Vp;
  const float*    v2 = path ? v2r : v2p;
  const float*    v1 = path ? v1r : v1p;
  const float*    u2 = path ? u2r : u2p;
  _Float16*       H  = path ? Hr  : Hp;

  // A staging map: 4 float4 per thread (128x32 f32 tile, fused v2 scale + cvt)
  const int a_colg = t & 7;       // *4 floats within the 32-wide K tile
  const int a_r0   = t >> 3;      // 0..31
  // B staging map: 2 b128 per thread (128x32 f16 tile)
  const int b_r0   = t >> 2;      // 0..63 (+64)
  const int b_colk = (t & 3) * 8;

  auto stageA = [&](int buf, int k0) {
    float4 v2v = reinterpret_cast<const float4*>(v2)[(k0 >> 2) + a_colg];
#pragma unroll
    for (int i = 0; i < 4; ++i) {
      int row = a_r0 + 32 * i;
      float4 xv = *reinterpret_cast<const float4*>(
          &x[(size_t)(m0 + row) * IN_F + k0 + a_colg * 4]);
      v4h hh;
      hh.x = (_Float16)(xv.x * v2v.x);
      hh.y = (_Float16)(xv.y * v2v.y);
      hh.z = (_Float16)(xv.z * v2v.z);
      hh.w = (_Float16)(xv.w * v2v.w);
      *reinterpret_cast<v4h*>(&As[buf][row][a_colg * 4]) = hh;
    }
  };
  auto stageB = [&](int buf, int k0) {
#pragma unroll
    for (int i = 0; i < 2; ++i) {
      int row = b_r0 + 64 * i;
      cp_b128(&W[(size_t)(s0 + row) * IN_F + k0 + b_colk], &Bs[buf][row][b_colk]);
    }
  };

  v8f acc[2][4] = {};
  const int NK = IN_F / BK;       // 128

  stageB(0, 0);
  stageA(0, 0);

  for (int it = 0; it < NK; ++it) {
    wait_async0();
    wait_ds0();
    __syncthreads();              // tile `it` is now visible to all waves
    if (it + 1 < NK) {            // fill the other buffer while we compute
      stageB((it + 1) & 1, (it + 1) * BK);
      stageA((it + 1) & 1, (it + 1) * BK);
      __builtin_prefetch(&x[(size_t)(m0 + a_r0) * IN_F + (it + 2) * BK], 0, 3);
    }
    const int cb = it & 1;

    v16h afr[2], bfr[4];
#pragma unroll
    for (int mi = 0; mi < 2; ++mi) {
      int row = waveM * 32 + mi * 16 + lr;
      int kf  = hlf * 8;          // elems 0..7: K kf.. ; elems 8..15: K kf+16..
      *reinterpret_cast<v8h*>(&afr[mi])       = *reinterpret_cast<const v8h*>(&As[cb][row][kf]);
      *(reinterpret_cast<v8h*>(&afr[mi]) + 1) = *reinterpret_cast<const v8h*>(&As[cb][row][kf + 16]);
    }
#pragma unroll
    for (int ni = 0; ni < 4; ++ni) {
      int row = waveN * 64 + ni * 16 + lr;    // N index
      int kf  = hlf * 8;                      // elems 0..7: K kf.. ; elems 8..15: K kf+16..
      *reinterpret_cast<v8h*>(&bfr[ni])       = *reinterpret_cast<const v8h*>(&Bs[cb][row][kf]);
      *(reinterpret_cast<v8h*>(&bfr[ni]) + 1) = *reinterpret_cast<const v8h*>(&Bs[cb][row][kf + 16]);
    }
#pragma unroll
    for (int mi = 0; mi < 2; ++mi)
#pragma unroll
      for (int ni = 0; ni < 4; ++ni)
        acc[mi][ni] = __builtin_amdgcn_wmma_f32_16x16x32_f16(
            false, afr[mi], false, bfr[ni], (short)0, acc[mi][ni], false, false);
  }

  // epilogue: apply per-split dequant scale, store f16 H
#pragma unroll
  for (int ni = 0; ni < 4; ++ni) {
    int s = s0 + waveN * 64 + ni * 16 + lr;
    float sc = v1[s] * u2[s];
#pragma unroll
    for (int mi = 0; mi < 2; ++mi) {
      int rbase = m0 + waveM * 32 + mi * 16 + hlf * 8;   // C layout: VGPR j -> M = 8*half + j
#pragma unroll
      for (int j = 0; j < 8; ++j)
        H[(size_t)(rbase + j) * SPLIT + s] = (_Float16)(acc[mi][ni][j] * sc);
    }
  }
}

// ---------------------------------------------------------------------------
// GEMM2: y[n,o] = u1[o]*Hp@sign(U)^T + u1R[o]*Hr@sign(U_R)^T + bias[o]
// All tile fills are pure copies -> async global->LDS when available.
// ---------------------------------------------------------------------------
__global__ __launch_bounds__(256) void gemm2_kernel(
    const _Float16* __restrict__ Hp, const _Float16* __restrict__ Hr,
    const _Float16* __restrict__ Up, const _Float16* __restrict__ Ur,
    const float* __restrict__ u1, const float* __restrict__ u1r,
    const float* __restrict__ bias, float* __restrict__ out)
{
  __shared__ _Float16 As[2][BM][BKP];
  __shared__ _Float16 Bs[2][BN][BKP];

  const int t    = threadIdx.x;
  const int lane = t & 31;
  const int wave = t >> 5;
  const int hlf  = lane >> 4;
  const int lr   = lane & 15;
  const int waveM = wave & 3;
  const int waveN = wave >> 2;
  const int m0 = blockIdx.y * BM;
  const int o0 = blockIdx.x * BN;

  // A / B staging maps: 2 b128 per thread each (128x32 f16 tiles)
  const int a_r0   = t >> 2;      // rows a_r0 and a_r0+64
  const int a_colk = (t & 3) * 8;
  const int b_r0   = t >> 2;
  const int b_colk = (t & 3) * 8;

  v8f acc[2][2][4] = {};          // [path][mi][ni]
  const int NK = SPLIT / BK;      // 32

#pragma unroll
  for (int path = 0; path < 2; ++path) {
    const _Float16* H = path ? Hr : Hp;
    const _Float16* W = path ? Ur : Up;

    auto stageA = [&](int buf, int k0) {
#pragma unroll
      for (int i = 0; i < 2; ++i) {
        int row = a_r0 + 64 * i;
        cp_b128(&H[(size_t)(m0 + row) * SPLIT + k0 + a_colk], &As[buf][row][a_colk]);
      }
    };
    auto stageB = [&](int buf, int k0) {
#pragma unroll
      for (int i = 0; i < 2; ++i) {
        int row = b_r0 + 64 * i;
        cp_b128(&W[(size_t)(o0 + row) * SPLIT + k0 + b_colk], &Bs[buf][row][b_colk]);
      }
    };

    __syncthreads();              // buffer handoff between paths
    stageA(0, 0);
    stageB(0, 0);

    for (int it = 0; it < NK; ++it) {
      wait_async0();
      wait_ds0();
      __syncthreads();
      if (it + 1 < NK) {
        stageA((it + 1) & 1, (it + 1) * BK);
        stageB((it + 1) & 1, (it + 1) * BK);
      }
      const int cb = it & 1;

      v16h afr[2], bfr[4];
#pragma unroll
      for (int mi = 0; mi < 2; ++mi) {
        int row = waveM * 32 + mi * 16 + lr;
        int kf  = hlf * 8;
        *reinterpret_cast<v8h*>(&afr[mi])       = *reinterpret_cast<const v8h*>(&As[cb][row][kf]);
        *(reinterpret_cast<v8h*>(&afr[mi]) + 1) = *reinterpret_cast<const v8h*>(&As[cb][row][kf + 16]);
      }
#pragma unroll
      for (int ni = 0; ni < 4; ++ni) {
        int row = waveN * 64 + ni * 16 + lr;
        int kf  = hlf * 8;
        *reinterpret_cast<v8h*>(&bfr[ni])       = *reinterpret_cast<const v8h*>(&Bs[cb][row][kf]);
        *(reinterpret_cast<v8h*>(&bfr[ni]) + 1) = *reinterpret_cast<const v8h*>(&Bs[cb][row][kf + 16]);
      }
#pragma unroll
      for (int mi = 0; mi < 2; ++mi)
#pragma unroll
        for (int ni = 0; ni < 4; ++ni)
          acc[path][mi][ni] = __builtin_amdgcn_wmma_f32_16x16x32_f16(
              false, afr[mi], false, bfr[ni], (short)0, acc[path][mi][ni], false, false);
    }
  }

  // epilogue: combine paths with f32 per-output scales + bias, store f32
#pragma unroll
  for (int ni = 0; ni < 4; ++ni) {
    int o = o0 + waveN * 64 + ni * 16 + lr;
    float s1 = u1[o], s1r = u1r[o], sb = bias[o];
#pragma unroll
    for (int mi = 0; mi < 2; ++mi) {
      int rbase = m0 + waveM * 32 + mi * 16 + hlf * 8;
#pragma unroll
      for (int j = 0; j < 8; ++j)
        out[(size_t)(rbase + j) * OUT_F + o] =
            acc[0][mi][ni][j] * s1 + acc[1][mi][ni][j] * s1r + sb;
    }
  }
}

// ---------------------------------------------------------------------------
extern "C" void kernel_launch(void* const* d_in, const int* in_sizes, int n_in,
                              void* d_out, int out_size, void* d_ws, size_t ws_size,
                              hipStream_t stream) {
  (void)in_sizes; (void)n_in; (void)out_size; (void)ws_size;

  const float* x    = (const float*)d_in[0];
  const float* V    = (const float*)d_in[1];
  const float* U    = (const float*)d_in[2];
  const float* v2   = (const float*)d_in[3];
  const float* v1   = (const float*)d_in[4];
  const float* u2   = (const float*)d_in[5];
  const float* u1   = (const float*)d_in[6];
  const float* V_R  = (const float*)d_in[7];
  const float* U_R  = (const float*)d_in[8];
  const float* v2R  = (const float*)d_in[9];
  const float* v1R  = (const float*)d_in[10];
  const float* u2R  = (const float*)d_in[11];
  const float* u1R  = (const float*)d_in[12];
  const float* bias = (const float*)d_in[13];
  float* out = (float*)d_out;

  char* ws = (char*)d_ws;
  const size_t WSZ = (size_t)SPLIT * IN_F * sizeof(_Float16);   // 8 MB per weight matrix
  _Float16* Vp16 = (_Float16*)(ws);
  _Float16* Up16 = (_Float16*)(ws + 1 * WSZ);
  _Float16* Vr16 = (_Float16*)(ws + 2 * WSZ);
  _Float16* Ur16 = (_Float16*)(ws + 3 * WSZ);
  _Float16* Hpm  = (_Float16*)(ws + 4 * WSZ);
  _Float16* Hrm  = (_Float16*)(ws + 4 * WSZ + (size_t)NTOT * SPLIT * sizeof(_Float16));

  const int n4 = SPLIT * IN_F / 4;   // 1,048,576 float4 groups per matrix
  pack_sign_f16<<<n4 / 256, 256, 0, stream>>>(V,   Vp16, n4);
  pack_sign_f16<<<n4 / 256, 256, 0, stream>>>(U,   Up16, n4);
  pack_sign_f16<<<n4 / 256, 256, 0, stream>>>(V_R, Vr16, n4);
  pack_sign_f16<<<n4 / 256, 256, 0, stream>>>(U_R, Ur16, n4);

  gemm1_kernel<<<dim3(SPLIT / BN, NTOT / BM, 2), 256, 0, stream>>>(
      x, Vp16, Vr16, v2, v2R, v1, v1R, u2, u2R, Hpm, Hrm);

  gemm2_kernel<<<dim3(OUT_F / BN, NTOT / BM), 256, 0, stream>>>(
      Hpm, Hrm, Up16, Ur16, u1, u1R, bias, out);
}
